// V_SSTGCN_27616639713350
// MI455X (gfx1250) — matrix operand order repaired and checked by reading, not compiled, not run on hardware
//
#include <hip/hip_runtime.h>
#include <hip/hip_bf16.h>
#include <math.h>

// ---------------------------------------------------------------------------
// Types for WMMA fragments (gfx1250, wave32)
// ---------------------------------------------------------------------------
typedef __attribute__((ext_vector_type(16))) _Float16 v16h;
typedef __attribute__((ext_vector_type(8)))  _Float16 h8v;
typedef __attribute__((ext_vector_type(8)))  float    v8f;

// Model constants
#define BB 64
#define NN 8
#define FF 16
#define IND 1280
#define OUTD 512
#define HD 640

// ---------------------------------------------------------------------------
// f32 -> f16 conversion
// ---------------------------------------------------------------------------
__global__ void k_f32_to_f16(const float* __restrict__ a, _Float16* __restrict__ o, size_t n) {
    size_t i = (size_t)blockIdx.x * blockDim.x + threadIdx.x;
    if (i < n) o[i] = (_Float16)a[i];
}

__global__ void k_zero_f32(float* __restrict__ p, size_t n) {
    size_t i = (size_t)blockIdx.x * blockDim.x + threadIdx.x;
    if (i < n) p[i] = 0.0f;
}

// ---------------------------------------------------------------------------
// Box coordinate MLP: Linear(4->128)+BN+ReLU, Linear(128->256)+BN+ReLU
// One block (128 threads) per row r = (b*N+n)*F + f; writes feats[:, 0:256]
// ---------------------------------------------------------------------------
__global__ void k_boxfeat(const float* __restrict__ box,
                          const float* __restrict__ c1W,
                          const float* __restrict__ g1, const float* __restrict__ b1,
                          const float* __restrict__ m1, const float* __restrict__ v1,
                          const float* __restrict__ c2W,
                          const float* __restrict__ g2, const float* __restrict__ b2,
                          const float* __restrict__ m2, const float* __restrict__ v2,
                          float* __restrict__ feats) {
    __shared__ float sf[128];
    int r = blockIdx.x;              // 0..8191
    int b = r >> 7;                  // / (N*F)
    int n = (r >> 4) & 7;            // (r/F)%N
    int f = r & 15;
    const float* bx = box + (((size_t)(b * FF + f) * NN) + n) * 4;
    float x0 = bx[0], x1 = bx[1], x2 = bx[2], x3 = bx[3];
    int t = threadIdx.x;             // 128 threads
    {
        const float* w = c1W + t * 4;
        float s = x0 * w[0] + x1 * w[1] + x2 * w[2] + x3 * w[3];
        s = (s - m1[t]) * rsqrtf(v1[t] + 1e-5f) * g1[t] + b1[t];
        sf[t] = s > 0.f ? s : 0.f;
    }
    __syncthreads();
    for (int o = t; o < 256; o += 128) {
        const float* w = c2W + (size_t)o * 128;
        float s = 0.f;
        #pragma unroll 8
        for (int k = 0; k < 128; ++k) s += sf[k] * w[k];
        s = (s - m2[o]) * rsqrtf(v2[o] + 1e-5f) * g2[o] + b2[o];
        feats[(size_t)r * IND + o] = s > 0.f ? s : 0.f;
    }
}

// ---------------------------------------------------------------------------
// Fragment helpers (ISA 16-bit A 16x32 / B 32x16 lane layouts, wave32):
//  A: lane L holds row M=L%16, K in [8*(L/16),+8) and [8*(L/16)+16,+8)
//  B: lane L holds col N=L%16, K in [8*(L/16),+8) and [8*(L/16)+16,+8)
// ---------------------------------------------------------------------------
__device__ __forceinline__ v16h frag_a_f16(const _Float16* __restrict__ A, size_t lda,
                                           int mrow, int kbase) {
    const h8v* p0 = (const h8v*)(A + (size_t)mrow * lda + kbase);
    const h8v* p1 = (const h8v*)(A + (size_t)mrow * lda + kbase + 16);
    h8v a0 = *p0, a1 = *p1;
    v16h a;
    #pragma unroll
    for (int j = 0; j < 8; ++j) { a[j] = a0[j]; a[j + 8] = a1[j]; }
    return a;
}

__device__ __forceinline__ v16h frag_b_f16(const _Float16* __restrict__ W, size_t ldw,
                                           int ncol, int kbase) {
    const h8v* p0 = (const h8v*)(W + (size_t)ncol * ldw + kbase);
    const h8v* p1 = (const h8v*)(W + (size_t)ncol * ldw + kbase + 16);
    h8v b0 = *p0, b1 = *p1;
    v16h b;
    #pragma unroll
    for (int j = 0; j < 8; ++j) { b[j] = b0[j]; b[j + 8] = b1[j]; }
    return b;
}

// ---------------------------------------------------------------------------
// Generic WMMA GEMM: C[M,N] = act(A16[M,K] * W16[N,K]^T + bias), f32 out.
// One wave computes a 16x64 tile (4 accumulators share one A fragment):
// 4 v_wmma_f32_16x16x32_f16 issues per A-fragment load.
// Requires M%16==0, N%64==0, K%32==0.
// ---------------------------------------------------------------------------
__global__ void k_gemm_f16(const _Float16* __restrict__ A, int lda,
                           const _Float16* __restrict__ W, int ldw,
                           const float* __restrict__ bias,
                           float* __restrict__ C, int ldc,
                           int M, int N, int K, int act) {
    int wave = blockIdx.x * (blockDim.x >> 5) + (threadIdx.x >> 5);
    int tilesN = N >> 6;                       // 64-wide tiles
    int total = (M >> 4) * tilesN;
    if (wave >= total) return;                 // wave-uniform: EXEC stays all-ones
    int tm = (wave / tilesN) << 4;
    int tn = (wave % tilesN) << 6;
    int lane = threadIdx.x & 31;
    int half = lane >> 4;
    int l16 = lane & 15;
    int mrow = tm + l16;
    v8f acc0 = {}, acc1 = {}, acc2 = {}, acc3 = {};
    for (int k0 = 0; k0 < K; k0 += 32) {
        if (k0 + 32 < K) {
            __builtin_prefetch(A + (size_t)mrow * lda + k0 + 32, 0, 1);
            __builtin_prefetch(W + (size_t)(tn + l16) * ldw + k0 + 32, 0, 1);
        }
        v16h a = frag_a_f16(A, lda, mrow, k0 + 8 * half);
        int kb = k0 + 8 * half;
        v16h b0 = frag_b_f16(W, ldw, tn + l16,      kb);
        v16h b1 = frag_b_f16(W, ldw, tn + 16 + l16, kb);
        v16h b2 = frag_b_f16(W, ldw, tn + 32 + l16, kb);
        v16h b3 = frag_b_f16(W, ldw, tn + 48 + l16, kb);
        acc0 = __builtin_amdgcn_wmma_f32_16x16x32_f16(false, a, false, b0, (short)0, acc0, false, false);
        acc1 = __builtin_amdgcn_wmma_f32_16x16x32_f16(false, a, false, b1, (short)0, acc1, false, false);
        acc2 = __builtin_amdgcn_wmma_f32_16x16x32_f16(false, a, false, b2, (short)0, acc2, false, false);
        acc3 = __builtin_amdgcn_wmma_f32_16x16x32_f16(false, a, false, b3, (short)0, acc3, false, false);
    }
    v8f accs[4] = {acc0, acc1, acc2, acc3};
    #pragma unroll
    for (int i = 0; i < 4; ++i) {
        #pragma unroll
        for (int r = 0; r < 8; ++r) {
            int row = tm + r + 8 * half;
            int col = tn + 16 * i + l16;
            float v = accs[i][r];
            if (bias) v += bias[col];
            if (act == 1) v = v > 0.f ? v : 0.f;
            C[(size_t)row * ldc + col] = v;
        }
    }
}

// ---------------------------------------------------------------------------
// RNN step (WMMA, f32 A converted in-register), 16x64 tile per wave:
//   hnext = tanh(hprev @ whh^T + pre[:, tin] + bhh); also scatter into xcat.
// M=512 (bn), N=K=640.
// ---------------------------------------------------------------------------
__global__ void k_rnn_step(const float* __restrict__ hprev,
                           const _Float16* __restrict__ whh16,
                           const float* __restrict__ pre,     // [8192,640], row = bn*F + t
                           const float* __restrict__ bhh,
                           float* __restrict__ hnext,
                           float* __restrict__ xcat,          // [8192,1280], row = bn*F + t
                           int tin, int tout, int colOff) {
    const int Mr = 512, Nn = HD, Kk = HD;
    int wave = blockIdx.x * (blockDim.x >> 5) + (threadIdx.x >> 5);
    int tilesN = Nn >> 6;                  // 10
    int total = (Mr >> 4) * tilesN;        // 320
    if (wave >= total) return;
    int tm = (wave / tilesN) << 4;
    int tn = (wave % tilesN) << 6;
    int lane = threadIdx.x & 31;
    int half = lane >> 4;
    int l16 = lane & 15;
    int m = tm + l16;
    v8f acc0 = {}, acc1 = {}, acc2 = {}, acc3 = {};
    for (int k0 = 0; k0 < Kk; k0 += 32) {
        v16h a;
        int ab = k0 + 8 * half;
        const float* pa = hprev + (size_t)m * Kk + ab;
        #pragma unroll
        for (int j = 0; j < 8; ++j) {
            a[j]     = (_Float16)pa[j];
            a[j + 8] = (_Float16)pa[j + 16];
        }
        int kb = k0 + 8 * half;
        v16h b0 = frag_b_f16(whh16, Kk, tn + l16,      kb);
        v16h b1 = frag_b_f16(whh16, Kk, tn + 16 + l16, kb);
        v16h b2 = frag_b_f16(whh16, Kk, tn + 32 + l16, kb);
        v16h b3 = frag_b_f16(whh16, Kk, tn + 48 + l16, kb);
        acc0 = __builtin_amdgcn_wmma_f32_16x16x32_f16(false, a, false, b0, (short)0, acc0, false, false);
        acc1 = __builtin_amdgcn_wmma_f32_16x16x32_f16(false, a, false, b1, (short)0, acc1, false, false);
        acc2 = __builtin_amdgcn_wmma_f32_16x16x32_f16(false, a, false, b2, (short)0, acc2, false, false);
        acc3 = __builtin_amdgcn_wmma_f32_16x16x32_f16(false, a, false, b3, (short)0, acc3, false, false);
    }
    v8f accs[4] = {acc0, acc1, acc2, acc3};
    #pragma unroll
    for (int i = 0; i < 4; ++i) {
        #pragma unroll
        for (int r = 0; r < 8; ++r) {
            int row = tm + r + 8 * half;
            int col = tn + 16 * i + l16;
            float v = accs[i][r] + pre[((size_t)row * FF + tin) * HD + col] + bhh[col];
            v = tanhf(v);
            hnext[(size_t)row * HD + col] = v;
            xcat[((size_t)row * FF + tout) * IND + colOff + col] = v;
        }
    }
}

// ---------------------------------------------------------------------------
// Reorder (b,n,f) -> node (b,f,n), add sinusoidal PE, emit f16 node features.
// ---------------------------------------------------------------------------
__global__ void k_reorder_pe(const float* __restrict__ xcat, _Float16* __restrict__ xg16) {
    size_t i = (size_t)blockIdx.x * blockDim.x + threadIdx.x;
    if (i >= (size_t)8192 * IND) return;
    int c = (int)(i % IND);
    int node = (int)(i / IND);
    int n = node & 7;
    int f = (node >> 3) & 15;
    int b = node >> 7;
    float v = xcat[(((size_t)(b * NN + n) * FF) + f) * IND + c];
    int pos = f * NN + n;
    int j2 = c & ~1;
    float ang = (float)pos * __expf(-logf(10000.0f) * (float)j2 / (float)IND);
    v += (c & 1) ? __cosf(ang) : __sinf(ang);
    xg16[i] = (_Float16)v;
}

// ---------------------------------------------------------------------------
// GAT attention scores el/er per node (h is f32 [nnodes,512])
// ---------------------------------------------------------------------------
__global__ void k_scores(const float* __restrict__ h, const float* __restrict__ al,
                         const float* __restrict__ ar, float* __restrict__ el,
                         float* __restrict__ er, int nnodes) {
    int nidx = blockIdx.x * blockDim.x + threadIdx.x;
    if (nidx >= nnodes) return;
    const float* hp = h + (size_t)nidx * OUTD;
    float a = 0.f, b = 0.f;
    for (int k = 0; k < OUTD; ++k) { float x = hp[k]; a += x * al[k]; b += x * ar[k]; }
    el[nidx] = a;
    er[nidx] = b;
}

// ---------------------------------------------------------------------------
// GAT edge-softmax + aggregation. dst is sorted; per-node edge range found by
// binary search. One block (128 threads) per node. Writes relu(out) into the
// concat buffer at column offset colOff (ldo = 1024).
// ---------------------------------------------------------------------------
__global__ void k_gat_aggr(const float* __restrict__ h, const float* __restrict__ el,
                           const float* __restrict__ er, const int* __restrict__ src,
                           const int* __restrict__ dst, int nedges,
                           float* __restrict__ out, int ldo, int colOff) {
    __shared__ float red[128];
    __shared__ int sLo, sHi;
    int node = blockIdx.x;
    int t = threadIdx.x;
    if (t == 0) {
        int lo = 0, hi = nedges;
        while (lo < hi) { int mid = (lo + hi) >> 1; if (dst[mid] < node) lo = mid + 1; else hi = mid; }
        sLo = lo;
        int lo2 = lo, hi2 = nedges;
        while (lo2 < hi2) { int mid = (lo2 + hi2) >> 1; if (dst[mid] <= node) lo2 = mid + 1; else hi2 = mid; }
        sHi = lo2;
    }
    __syncthreads();
    int lo = sLo, hi = sHi;
    float ern = er[node];
    float mx = -3.0e38f;
    for (int e = lo + t; e < hi; e += 128) {
        float x = el[src[e]] + ern;
        x = x > 0.f ? x : 0.2f * x;
        mx = fmaxf(mx, x);
    }
    red[t] = mx;
    __syncthreads();
    for (int s = 64; s > 0; s >>= 1) { if (t < s) red[t] = fmaxf(red[t], red[t + s]); __syncthreads(); }
    mx = red[0];
    __syncthreads();
    float sum = 0.f;
    for (int e = lo + t; e < hi; e += 128) {
        float x = el[src[e]] + ern;
        x = x > 0.f ? x : 0.2f * x;
        sum += __expf(x - mx);
    }
    red[t] = sum;
    __syncthreads();
    for (int s = 64; s > 0; s >>= 1) { if (t < s) red[t] += red[t + s]; __syncthreads(); }
    sum = red[0];
    float inv = (hi > lo) ? 1.0f / sum : 0.0f;
    float acc[4] = {0.f, 0.f, 0.f, 0.f};
    for (int e = lo; e < hi; ++e) {
        int sidx = src[e];
        float x = el[sidx] + ern;
        x = x > 0.f ? x : 0.2f * x;
        float w = __expf(x - mx) * inv;
        const float* hp = h + (size_t)sidx * OUTD;
        #pragma unroll
        for (int j = 0; j < 4; ++j) acc[j] += w * hp[t + 128 * j];
    }
    #pragma unroll
    for (int j = 0; j < 4; ++j) {
        float v = acc[j];
        v = v > 0.f ? v : 0.f;
        out[(size_t)node * ldo + colOff + t + 128 * j] = v;
    }
}

// ---------------------------------------------------------------------------
// Gather human rows (n==0) from x3 [8192,1024] -> human16 [1024,1024] f16
// ---------------------------------------------------------------------------
__global__ void k_gather_human16(const float* __restrict__ x3, _Float16* __restrict__ human16) {
    size_t i = (size_t)blockIdx.x * blockDim.x + threadIdx.x;
    if (i >= (size_t)1024 * 1024) return;
    int c = (int)(i & 1023);
    int m = (int)(i >> 10);
    human16[i] = (_Float16)x3[((size_t)m * NN) * 1024 + c];
}

// ---------------------------------------------------------------------------
// Final classifier layer (N=174 not WMMA-tileable) + sum over frames.
// out[b,c] = sum_f (h2[b*F+f] . W3[c] + b3[c])
// ---------------------------------------------------------------------------
__global__ void k_cls_final(const float* __restrict__ h2, const float* __restrict__ W3,
                            const float* __restrict__ b3, float* __restrict__ out) {
    int i = blockIdx.x * blockDim.x + threadIdx.x;
    if (i >= BB * 174) return;
    int c = i % 174;
    int b = i / 174;
    const float* w = W3 + (size_t)c * 512;
    float s = 0.f;
    for (int f = 0; f < FF; ++f) {
        const float* hp = h2 + ((size_t)(b * FF + f)) * 512;
        float d = 0.f;
        for (int k = 0; k < 512; ++k) d += hp[k] * w[k];
        s += d + b3[c];
    }
    out[(size_t)b * 174 + c] = s;
}

// ---------------------------------------------------------------------------
// Host-side orchestration
// ---------------------------------------------------------------------------
static inline int cdiv(long long a, long long b) { return (int)((a + b - 1) / b); }

extern "C" void kernel_launch(void* const* d_in, const int* in_sizes, int n_in,
                              void* d_out, int out_size, void* d_ws, size_t ws_size,
                              hipStream_t stream) {
    // ---- inputs (setup_inputs order) ----
    const float* node_features = (const float*)d_in[1];   // [B,N,F,2048]
    const float* box_input     = (const float*)d_in[3];   // [B,F,N,4]
    const float* app_W = (const float*)d_in[4];   const float* app_b = (const float*)d_in[5];
    const float* c1_W  = (const float*)d_in[6];
    const float* bn1_g = (const float*)d_in[7];   const float* bn1_b = (const float*)d_in[8];
    const float* bn1_m = (const float*)d_in[9];   const float* bn1_v = (const float*)d_in[10];
    const float* c2_W  = (const float*)d_in[11];
    const float* bn2_g = (const float*)d_in[12];  const float* bn2_b = (const float*)d_in[13];
    const float* bn2_m = (const float*)d_in[14];  const float* bn2_v = (const float*)d_in[15];
    const float* wih_f = (const float*)d_in[16];  const float* whh_f = (const float*)d_in[17];
    const float* bih_f = (const float*)d_in[18];  const float* bhh_f = (const float*)d_in[19];
    const float* wih_b = (const float*)d_in[20];  const float* whh_b = (const float*)d_in[21];
    const float* bih_b = (const float*)d_in[22];  const float* bhh_b = (const float*)d_in[23];
    const float* sp1_W = (const float*)d_in[24];  const float* sp1_al = (const float*)d_in[25];
    const float* sp1_ar = (const float*)d_in[26];
    const float* in1_W = (const float*)d_in[27];  const float* in1_al = (const float*)d_in[28];
    const float* in1_ar = (const float*)d_in[29];
    const float* sp2_W = (const float*)d_in[30];  const float* sp2_al = (const float*)d_in[31];
    const float* sp2_ar = (const float*)d_in[32];
    const float* in2_W = (const float*)d_in[33];  const float* in2_al = (const float*)d_in[34];
    const float* in2_ar = (const float*)d_in[35];
    const float* cls_W1 = (const float*)d_in[36]; const float* cls_b1 = (const float*)d_in[37];
    const float* cls_W2 = (const float*)d_in[38]; const float* cls_b2 = (const float*)d_in[39];
    const float* cls_W3 = (const float*)d_in[40]; const float* cls_b3 = (const float*)d_in[41];
    const int* sp_src = (const int*)d_in[42];  const int* sp_dst = (const int*)d_in[43];
    const int* in_src = (const int*)d_in[44];  const int* in_dst = (const int*)d_in[45];
    const int nsp = in_sizes[42];
    const int nin = in_sizes[44];

    float* out = (float*)d_out;
    char* ws = (char*)d_ws;
    const size_t MBy = 1ull << 20;

    // ---- workspace layout with manual region reuse ----
    // Region A (0..48MB): nf16 -> xcat(f32) -> x3(f32)
    _Float16* nf16 = (_Float16*)(ws + 0);
    float* xcat = (float*)(ws + 0);
    float* x3   = (float*)(ws + 0);
    // Region B (48..96MB): feats(f32) -> pre_f/pre_b -> x2(f32)
    float* feats = (float*)(ws + 48 * MBy);
    float* pre_f = (float*)(ws + 48 * MBy);
    float* pre_b = (float*)(ws + 70 * MBy);
    float* x2    = (float*)(ws + 48 * MBy);
    // Region C (96..120MB): feats16 -> xg16 -> x2_16
    _Float16* feats16 = (_Float16*)(ws + 96 * MBy);
    _Float16* xg16    = (_Float16*)(ws + 96 * MBy);
    _Float16* x2_16   = (_Float16*)(ws + 96 * MBy);
    // GAT projection buffer
    float* hgat = (float*)(ws + 120 * MBy);          // [8192,512] f32
    float* el   = (float*)(ws + 137 * MBy);
    float* er   = (float*)(ws + 137 * MBy + 64 * 1024);
    float* hA   = (float*)(ws + 138 * MBy);          // [512,640]
    float* hB   = (float*)(ws + 140 * MBy);
    _Float16* appW16 = (_Float16*)(ws + 142 * MBy);
    _Float16* wihf16 = (_Float16*)(ws + 147 * MBy);
    _Float16* wihb16 = (_Float16*)(ws + 149 * MBy);
    _Float16* whhf16 = (_Float16*)(ws + 151 * MBy);
    _Float16* whhb16 = (_Float16*)(ws + 152 * MBy);
    _Float16* sp1W16 = (_Float16*)(ws + 153 * MBy);
    _Float16* in1W16 = (_Float16*)(ws + 155 * MBy);
    _Float16* sp2W16 = (_Float16*)(ws + 157 * MBy);
    _Float16* in2W16 = (_Float16*)(ws + 159 * MBy);
    _Float16* cW1_16 = (_Float16*)(ws + 161 * MBy);
    _Float16* cW2_16 = (_Float16*)(ws + 164 * MBy);
    _Float16* human16 = (_Float16*)(ws + 166 * MBy);
    float* h1 = (float*)(ws + 169 * MBy);
    _Float16* h1_16 = (_Float16*)(ws + 174 * MBy);
    float* h2 = (float*)(ws + 177 * MBy);

    const int nnodes = BB * FF * NN;                 // 8192

    auto conv = [&](const float* a, _Float16* o, size_t n) {
        k_f32_to_f16<<<cdiv((long long)n, 256), 256, 0, stream>>>(a, o, n);
    };
    auto gemm = [&](const _Float16* A, int lda, const _Float16* W, int ldw,
                    const float* bias, float* C, int ldc, int M, int N, int K, int act) {
        int tiles = (M >> 4) * (N >> 6);     // 16x64 tiles per wave
        k_gemm_f16<<<cdiv(tiles, 8), 256, 0, stream>>>(A, lda, W, ldw, bias, C, ldc, M, N, K, act);
    };

    // 1) box spatial features -> feats[:, 0:256]
    k_boxfeat<<<nnodes, 128, 0, stream>>>(box_input, c1_W, bn1_g, bn1_b, bn1_m, bn1_v,
                                          c2_W, bn2_g, bn2_b, bn2_m, bn2_v, feats);
    // 2) appearance features -> feats[:, 256:1280]  (WMMA GEMM 8192x1024x2048)
    conv(node_features, nf16, (size_t)8192 * 2048);
    conv(app_W, appW16, (size_t)1024 * 2048);
    gemm(nf16, 2048, appW16, 2048, app_b, feats + 256, IND, 8192, 1024, 2048, 0);
    // 3) feats -> f16
    conv(feats, feats16, (size_t)8192 * IND);
    // 4) RNN input projections (8192x640x1280 each)
    conv(wih_f, wihf16, (size_t)HD * IND);
    conv(wih_b, wihb16, (size_t)HD * IND);
    conv(whh_f, whhf16, (size_t)HD * HD);
    conv(whh_b, whhb16, (size_t)HD * HD);
    gemm(feats16, IND, wihf16, IND, bih_f, pre_f, HD, 8192, HD, IND, 0);
    gemm(feats16, IND, wihb16, IND, bih_b, pre_b, HD, 8192, HD, IND, 0);
    // 5) sequential RNN scans (WMMA per step)
    {
        const int tiles = (512 >> 4) * (HD >> 6);    // 32*10 = 320 waves
        const int blks = cdiv(tiles, 8);
        k_zero_f32<<<cdiv((long long)512 * HD, 256), 256, 0, stream>>>(hA, (size_t)512 * HD);
        float* hp = hA; float* hn = hB;
        for (int t = 0; t < FF; ++t) {
            k_rnn_step<<<blks, 256, 0, stream>>>(hp, whhf16, pre_f, bhh_f, hn, xcat, t, t, 0);
            float* tmp = hp; hp = hn; hn = tmp;
        }
        k_zero_f32<<<cdiv((long long)512 * HD, 256), 256, 0, stream>>>(hp, (size_t)512 * HD);
        for (int t = 0; t < FF; ++t) {
            int tr = FF - 1 - t;
            k_rnn_step<<<blks, 256, 0, stream>>>(hp, whhb16, pre_b, bhh_b, hn, xcat, tr, tr, HD);
            float* tmp = hp; hp = hn; hn = tmp;
        }
    }
    // 6) reorder to node layout + positional embedding -> xg16
    k_reorder_pe<<<cdiv((long long)nnodes * IND, 256), 256, 0, stream>>>(xcat, xg16);
    // 7) convert GAT / classifier weights
    conv(sp1_W, sp1W16, (size_t)OUTD * IND);
    conv(in1_W, in1W16, (size_t)OUTD * IND);
    conv(sp2_W, sp2W16, (size_t)OUTD * 1024);
    conv(in2_W, in2W16, (size_t)OUTD * 1024);
    conv(cls_W1, cW1_16, (size_t)1024 * 1024);
    conv(cls_W2, cW2_16, (size_t)512 * 1024);
    // 8) SSTGC block 1: spatial then inner GAT -> x2 [8192,1024]
    gemm(xg16, IND, sp1W16, IND, nullptr, hgat, OUTD, nnodes, OUTD, IND, 0);
    k_scores<<<cdiv(nnodes, 256), 256, 0, stream>>>(hgat, sp1_al, sp1_ar, el, er, nnodes);
    k_gat_aggr<<<nnodes, 128, 0, stream>>>(hgat, el, er, sp_src, sp_dst, nsp, x2, 1024, 0);
    gemm(xg16, IND, in1W16, IND, nullptr, hgat, OUTD, nnodes, OUTD, IND, 0);
    k_scores<<<cdiv(nnodes, 256), 256, 0, stream>>>(hgat, in1_al, in1_ar, el, er, nnodes);
    k_gat_aggr<<<nnodes, 128, 0, stream>>>(hgat, el, er, in_src, in_dst, nin, x2, 1024, OUTD);
    // 9) SSTGC block 2 -> x3 [8192,1024]
    conv(x2, x2_16, (size_t)nnodes * 1024);
    gemm(x2_16, 1024, sp2W16, 1024, nullptr, hgat, OUTD, nnodes, OUTD, 1024, 0);
    k_scores<<<cdiv(nnodes, 256), 256, 0, stream>>>(hgat, sp2_al, sp2_ar, el, er, nnodes);
    k_gat_aggr<<<nnodes, 128, 0, stream>>>(hgat, el, er, sp_src, sp_dst, nsp, x3, 1024, 0);
    gemm(x2_16, 1024, in2W16, 1024, nullptr, hgat, OUTD, nnodes, OUTD, 1024, 0);
    k_scores<<<cdiv(nnodes, 256), 256, 0, stream>>>(hgat, in2_al, in2_ar, el, er, nnodes);
    k_gat_aggr<<<nnodes, 128, 0, stream>>>(hgat, el, er, in_src, in_dst, nin, x3, 1024, OUTD);
    // 10) classifier on human nodes
    k_gather_human16<<<cdiv((long long)1024 * 1024, 256), 256, 0, stream>>>(x3, human16);
    gemm(human16, 1024, cW1_16, 1024, cls_b1, h1, 1024, 1024, 1024, 1024, 1);
    conv(h1, h1_16, (size_t)1024 * 1024);
    gemm(h1_16, 1024, cW2_16, 1024, cls_b2, h2, 512, 1024, 512, 1024, 1);
    k_cls_final<<<cdiv(BB * 174, 256), 256, 0, stream>>>(h2, cls_W3, cls_b3, out);
}
